// CharRNN_86792699118065
// MI455X (gfx1250) — matrix operand order repaired and checked by reading, not compiled, not run on hardware
//
#include <hip/hip_runtime.h>
#include <hip/hip_bf16.h>
#include <math.h>

// Problem dims (CharRNN): x[B,L], emb[V,E], w_e[E,H], w_h[H,H], w_o[H,V]
#define BB 64
#define LL 512
#define VV 256
#define EE 512
#define HH 1024

typedef __attribute__((ext_vector_type(16))) __bf16 v16bf;
typedef __attribute__((ext_vector_type(8)))  __bf16 v8bf;
typedef __attribute__((ext_vector_type(8)))  float  v8f;

// ---- WMMA helpers -----------------------------------------------------------
// A-operand (16-bit, 16x32): lane holds row M = lane&15, half = lane>>4.
// This code puts K = half*16 + i in element i (B matches)  ->  two contiguous
// 8-element (16B) chunks at [k + half*16] and [k + half*16 + 8].
static __device__ __forceinline__ v16bf load_a16(const __bf16* __restrict__ row,
                                                 int k, int half) {
  v8bf lo = *(const v8bf*)(row + k + half * 16);
  v8bf hi = *(const v8bf*)(row + k + half * 16 + 8);
  return __builtin_shufflevector(lo, hi, 0,1,2,3,4,5,6,7,8,9,10,11,12,13,14,15);
}

// B-operand (16-bit, 32x16): lane holds column N = lane&15; this code uses
// K = half*16 + i in element i, matching A. With B stored transposed ([N,K] row-major) this is one
// contiguous 32B load per lane.
static __device__ __forceinline__ v16bf load_bT16(const __bf16* __restrict__ colRow,
                                                  int k, int half) {
  return *(const v16bf*)(colRow + k + half * 16);
}

static __device__ __forceinline__ v8f wmma_bf16(v16bf a, v16bf b, v8f c) {
  return __builtin_amdgcn_wmma_f32_16x16x32_bf16(false, a, false, b,
                                                 (short)0, c, false, false);
}

// ---- Prep kernels -----------------------------------------------------------
__global__ void f32_to_bf16_kernel(const float* __restrict__ src,
                                   __bf16* __restrict__ dst, int n) {
  int i = blockIdx.x * blockDim.x + threadIdx.x;
  if (i < n) dst[i] = (__bf16)src[i];
}

// src [rows, cols] f32 -> dst [cols, rows] bf16
__global__ void transpose_to_bf16_kernel(const float* __restrict__ src,
                                         __bf16* __restrict__ dst,
                                         int rows, int cols) {
  int i = blockIdx.x * blockDim.x + threadIdx.x;
  if (i < rows * cols) {
    int r = i / cols, c = i % cols;
    dst[c * rows + r] = (__bf16)src[i];
  }
}

__global__ void zero_i32_kernel(int* __restrict__ p, int n) {
  int i = blockIdx.x * blockDim.x + threadIdx.x;
  if (i < n) p[i] = 0;
}

// ---- x_proj: xproj[l*B+b, :] = emb[x[b,l]] @ w_e  (f32 out) ----------------
__global__ void xproj_kernel(const int* __restrict__ x,
                             const __bf16* __restrict__ emb16,   // [V,E]
                             const __bf16* __restrict__ weT,     // [H,E]
                             float* __restrict__ xproj) {        // [L*B,H]
  int gtid = blockIdx.x * 256 + threadIdx.x;
  int wave = gtid >> 5;
  int lane = gtid & 31;
  int half = lane >> 4;
  int col  = lane & 15;

  int mTile  = wave >> 4;        // 0 .. (L*B/16)-1  = 2047
  int nBase  = (wave & 15) * 64; // 16 groups of 64 cols

  int m   = mTile * 16 + (lane & 15);   // global row  (= l*B + b)
  int l   = m / BB;
  int b   = m % BB;
  int tok = x[b * LL + l];
  const __bf16* aRow = emb16 + (size_t)tok * EE;

  v8f z = {0.f,0.f,0.f,0.f,0.f,0.f,0.f,0.f};
  v8f c[4] = {z, z, z, z};

  for (int k = 0; k < EE; k += 32) {
    v16bf a = load_a16(aRow, k, half);
#pragma unroll
    for (int t = 0; t < 4; ++t) {
      v16bf bm = load_bT16(weT + (size_t)(nBase + t * 16 + col) * EE, k, half);
      c[t] = wmma_bf16(a, bm, c[t]);
    }
  }

#pragma unroll
  for (int t = 0; t < 4; ++t)
#pragma unroll
    for (int p = 0; p < 8; ++p) {
      int r = mTile * 16 + p + half * 8;               // C row = p + half*8
      xproj[(size_t)r * HH + nBase + t * 16 + col] = c[t][p];
    }
}

// ---- Persistent recurrence: h_l = tanh(h_{l-1} @ w_h + xproj_l) ------------
// 64 workgroups, each owns a 16-column slice of w_h^T staged once into LDS
// (32KB of the 320KB/WGP). Loop over all 512 steps inside one kernel; per-step
// grid sync via per-step agent-scope atomic counters (no launch overhead on the
// serial chain, weights never leave LDS).
__global__ void rnn_persistent_kernel(const __bf16* __restrict__ h0,     // [B,H]
                                      const __bf16* __restrict__ whT,    // [H,H]
                                      const float*  __restrict__ xproj,  // [L*B,H]
                                      __bf16* __restrict__ hs16,         // [L*B,H]
                                      float*  __restrict__ final_h,      // [B,H]
                                      int*    __restrict__ step_ctr) {   // [L]
  __shared__ __align__(32) __bf16 lds_b[16 * HH];   // 32KB: 16 rows of w_h^T
  __shared__ float red[4][2][256];                  // 8KB partial tiles

  const int tid   = threadIdx.x;
  const int nTile = blockIdx.x;                     // 0..63 (H/16)

  // Stage this block's B slice into LDS (one time).
  {
    const uint4* src = (const uint4*)(whT + (size_t)nTile * 16 * HH);
    uint4* dst = (uint4*)lds_b;
#pragma unroll
    for (int i = 0; i < 8; ++i) dst[tid + i * 256] = src[tid + i * 256];
  }
  __syncthreads();

  const int wave  = tid >> 5;
  const int lane  = tid & 31;
  const int half  = lane >> 4;
  const int col   = lane & 15;
  const int mTile = wave & 3;       // 0..3  (B/16)
  const int kHalf = wave >> 2;      // 0..1  K split
  const int kOff  = kHalf * 512;

  const __bf16* ldsBcol = lds_b + (size_t)col * HH + kOff;

  for (int l = 0; l < LL; ++l) {
    const __bf16* hprev = (l == 0) ? h0 : hs16 + (size_t)(l - 1) * (BB * HH);
    const __bf16* aRow  = hprev + (size_t)(mTile * 16 + (lane & 15)) * HH + kOff;

    v8f c = {0.f,0.f,0.f,0.f,0.f,0.f,0.f,0.f};
#pragma unroll
    for (int k = 0; k < 512; k += 32) {
      v16bf a  = load_a16(aRow, k, half);                 // global (h_{l-1})
      v16bf bm = *(const v16bf*)(ldsBcol + k + half * 16); // LDS   (w_h^T)
      c = wmma_bf16(a, bm, c);
    }

#pragma unroll
    for (int p = 0; p < 8; ++p)
      red[mTile][kHalf][(p + half * 8) * 16 + col] = c[p];
    __syncthreads();

    // 256 threads reduce the two K-partials of all 4 tiles (4 outputs each).
    __bf16* hout = hs16 + (size_t)l * (BB * HH);
    const float* xl = xproj + (size_t)l * (BB * HH);
    const int r  = tid >> 4;
    const int cc = tid & 15;
#pragma unroll
    for (int mt = 0; mt < 4; ++mt) {
      float s = red[mt][0][tid] + red[mt][1][tid];
      int bi = mt * 16 + r;                // batch index
      int hi = nTile * 16 + cc;            // hidden index
      s += xl[(size_t)bi * HH + hi];
      s = tanhf(s);
      hout[(size_t)bi * HH + hi] = (__bf16)s;
      if (l == LL - 1) final_h[(size_t)bi * HH + hi] = s;
    }

    // Grid-wide step barrier: release our h slice, wait for all 64 blocks.
    __threadfence();
    __syncthreads();
    if (tid == 0) {
      __hip_atomic_fetch_add(step_ctr + l, 1, __ATOMIC_RELEASE,
                             __HIP_MEMORY_SCOPE_AGENT);
      while (__hip_atomic_load(step_ctr + l, __ATOMIC_ACQUIRE,
                               __HIP_MEMORY_SCOPE_AGENT) < 64) {
        __builtin_amdgcn_s_sleep(1);
      }
    }
    __syncthreads();
    __threadfence();
  }
}

// ---- Output projection: logits[b,l,:] = hs[l,b,:] @ w_o --------------------
__global__ void out_kernel(const __bf16* __restrict__ hs16,  // [L*B,H]
                           const __bf16* __restrict__ woT,   // [V,H]
                           float* __restrict__ logits) {     // [B,L,V]
  int gtid = blockIdx.x * 256 + threadIdx.x;
  int wave = gtid >> 5;
  int lane = gtid & 31;
  int half = lane >> 4;
  int col  = lane & 15;

  int mTile = wave >> 4;       // 0..2047  (L*B/16)
  int nTile = wave & 15;       // 0..15    (V/16)

  const __bf16* aRow = hs16 + (size_t)(mTile * 16 + (lane & 15)) * HH;
  const __bf16* bCol = woT  + (size_t)(nTile * 16 + col) * HH;

  v8f c = {0.f,0.f,0.f,0.f,0.f,0.f,0.f,0.f};
#pragma unroll
  for (int k = 0; k < HH; k += 32) {
    v16bf a  = load_a16(aRow, k, half);
    v16bf bm = load_bT16(bCol, k, half);
    c = wmma_bf16(a, bm, c);
  }

#pragma unroll
  for (int p = 0; p < 8; ++p) {
    int m = mTile * 16 + p + half * 8;   // = l*B + b
    int l = m / BB;
    int b = m % BB;
    int v = nTile * 16 + col;
    logits[(size_t)b * (LL * VV) + (size_t)l * VV + v] = c[p];
  }
}

// ---- Host launcher ----------------------------------------------------------
extern "C" void kernel_launch(void* const* d_in, const int* in_sizes, int n_in,
                              void* d_out, int out_size, void* d_ws, size_t ws_size,
                              hipStream_t stream) {
  (void)in_sizes; (void)n_in; (void)out_size; (void)ws_size;

  const int*   x      = (const int*)  d_in[0];
  const float* hidden = (const float*)d_in[1];
  const float* emb    = (const float*)d_in[2];
  const float* w_e    = (const float*)d_in[3];
  const float* w_h    = (const float*)d_in[4];
  const float* w_o    = (const float*)d_in[5];

  float* logits  = (float*)d_out;
  float* final_h = logits + (size_t)BB * LL * VV;

  // Workspace carve-out (256B aligned slices)
  char*  ws  = (char*)d_ws;
  size_t off = 0;
  auto carve = [&](size_t bytes) -> void* {
    void* p = ws + off;
    off += (bytes + 255) & ~(size_t)255;
    return p;
  };
  __bf16* emb16 = (__bf16*)carve((size_t)VV * EE * 2);
  __bf16* weT   = (__bf16*)carve((size_t)HH * EE * 2);
  __bf16* whT   = (__bf16*)carve((size_t)HH * HH * 2);
  __bf16* woT   = (__bf16*)carve((size_t)VV * HH * 2);
  __bf16* h0    = (__bf16*)carve((size_t)BB * HH * 2);
  int*    ctr   = (int*)   carve((size_t)LL * 4);
  float*  xproj = (float*) carve((size_t)LL * BB * HH * 4);
  __bf16* hs16  = (__bf16*)carve((size_t)LL * BB * HH * 2);

  // 1) Convert / transpose weights to bf16; zero step-barrier counters
  //    (re-zeroed every call so graph replays stay deterministic).
  f32_to_bf16_kernel<<<(VV * EE + 255) / 256, 256, 0, stream>>>(emb, emb16, VV * EE);
  transpose_to_bf16_kernel<<<(EE * HH + 255) / 256, 256, 0, stream>>>(w_e, weT, EE, HH);
  transpose_to_bf16_kernel<<<(HH * HH + 255) / 256, 256, 0, stream>>>(w_h, whT, HH, HH);
  transpose_to_bf16_kernel<<<(HH * VV + 255) / 256, 256, 0, stream>>>(w_o, woT, HH, VV);
  f32_to_bf16_kernel<<<(BB * HH + 255) / 256, 256, 0, stream>>>(hidden, h0, BB * HH);
  zero_i32_kernel<<<(LL + 255) / 256, 256, 0, stream>>>(ctr, LL);

  // 2) x_proj = emb[x] @ w_e  (34 GFLOP, fully parallel WMMA)
  xproj_kernel<<<4096, 256, 0, stream>>>(x, emb16, weT, xproj);

  // 3) Recurrence: one persistent kernel, 64 co-resident blocks, w_h in LDS,
  //    per-step grid barrier on the serial chain instead of 512 launches.
  rnn_persistent_kernel<<<64, 256, 0, stream>>>(h0, whT, xproj, hs16, final_h, ctr);

  // 4) logits = hs @ w_o  (17 GFLOP, fully parallel WMMA)
  out_kernel<<<4096, 256, 0, stream>>>(hs16, woT, logits);
}
